// Network_41858751267393
// MI455X (gfx1250) — matrix operand order repaired and checked by reading, not compiled, not run on hardware
//
#include <hip/hip_runtime.h>
#include <hip/hip_bf16.h>

typedef __attribute__((ext_vector_type(16))) _Float16 v16h;
typedef __attribute__((ext_vector_type(8)))  float    v8f;

union FragU { float4 f4[2]; v16h h; };
union H8   { float4 f; _Float16 h[8]; };
union H4   { float2 f; _Float16 h[4]; };

#define BM   128
#define BN   256
#define BK   32
#define LSTR 40                 // halves; 80B row stride -> conflict-free b128 frag reads
#define ABUF (BM * LSTR)        // 5120 halves
#define BBUF (BN * LSTR)        // 10240 halves
#define BUFSZ (ABUF + BBUF)     // 15360 halves per buffer

// ---------------------------------------------------------------------------
// WMMA GEMM, double-buffered LDS with CDNA5 async global->LDS loads.
// C[M,N] = A[M,K] * W[K,N] (+bias). A f16 [M x lda], Wt f16 [N x ldw]
// (weights pre-transposed so K is contiguous for both operands).
// ---------------------------------------------------------------------------
template<bool OUT16, bool HASBIAS>
__global__ __launch_bounds__(256)
void k_gemm(const _Float16* __restrict__ A, int lda,
            const _Float16* __restrict__ Wt, int ldw,
            const float* __restrict__ bias,
            _Float16* __restrict__ out16, float* __restrict__ out32,
            int M, int N, int K, int ldo)
{
    __shared__ _Float16 lds[2 * BUFSZ + 8];   // [buf0 A|B][buf1 A|B][16B dummy]

    const int tid  = threadIdx.x;
    const int lane = tid & 31;
    const int wv   = tid >> 5;
    const int wm   = wv & 1;            // 2 waves along M, 64 rows each
    const int wn   = wv >> 1;           // 4 waves along N, 64 cols each
    const int hl   = lane >> 4;
    const int lid  = lane & 15;
    const int m0   = blockIdx.x * BM;
    const int n0   = blockIdx.y * BN;

    // pre-zero LDS: OOB guard rows are never overwritten -> stay zero
    {
        float4 z = make_float4(0.f, 0.f, 0.f, 0.f);
        float4* p = (float4*)lds;
        for (int i = tid; i < (2 * BUFSZ + 8) / 8; i += 256) p[i] = z;
    }
    __syncthreads();

    const uint32_t dummy = (uint32_t)(uintptr_t)(lds + 2 * BUFSZ);

    // loader state: 2 A-chunks + 4 B-chunks of 8 halves per thread
    const _Float16* ga[2]; uint32_t la[2][2];
    const _Float16* gb[4]; uint32_t lb[4][2];
#pragma unroll
    for (int r = 0; r < 2; ++r) {
        int q = tid + r * 256;
        int row = q >> 2, kc = (q & 3) << 3;
        int gr = m0 + row;
        bool ok = gr < M;
        ga[r] = A + (size_t)(ok ? gr : (M - 1)) * lda + kc;
        uint32_t rel = (uint32_t)(row * LSTR + kc);
        la[r][0] = ok ? (uint32_t)(uintptr_t)(lds + rel) : dummy;
        la[r][1] = ok ? (uint32_t)(uintptr_t)(lds + BUFSZ + rel) : dummy;
    }
#pragma unroll
    for (int r = 0; r < 4; ++r) {
        int q = tid + r * 256;
        int row = q >> 2, kc = (q & 3) << 3;
        int gn = n0 + row;
        bool ok = gn < N;
        gb[r] = Wt + (size_t)(ok ? gn : (N - 1)) * ldw + kc;
        uint32_t rel = (uint32_t)(ABUF + row * LSTR + kc);
        lb[r][0] = ok ? (uint32_t)(uintptr_t)(lds + rel) : dummy;
        lb[r][1] = ok ? (uint32_t)(uintptr_t)(lds + BUFSZ + rel) : dummy;
    }

    v8f acc[4][4];
#pragma unroll
    for (int mi = 0; mi < 4; ++mi)
#pragma unroll
        for (int ni = 0; ni < 4; ++ni) acc[mi][ni] = (v8f)(0.f);

    // b must always be a compile-time constant at the call sites
    auto issue = [&](int b, int koff) {
#pragma unroll
        for (int r = 0; r < 2; ++r)
            asm volatile("global_load_async_to_lds_b128 %0, %1, off"
                         :: "v"(la[r][b]), "v"(ga[r] + koff) : "memory");
#pragma unroll
        for (int r = 0; r < 4; ++r)
            asm volatile("global_load_async_to_lds_b128 %0, %1, off"
                         :: "v"(lb[r][b]), "v"(gb[r] + koff) : "memory");
    };
    auto asyncwait = [&]() {
        asm volatile("s_wait_asynccnt 0x0" ::: "memory");
    };
    auto compute = [&](int b) {
        const _Float16* As = lds + b * BUFSZ;
        const _Float16* Bs = As + ABUF;
        v16h af[4], bf[4];
#pragma unroll
        for (int mi = 0; mi < 4; ++mi) {
            int row = wm * 64 + mi * 16 + lid;
            FragU u;
            u.f4[0] = *(const float4*)(As + row * LSTR + hl * 8);       // K 0-7 / 8-15
            u.f4[1] = *(const float4*)(As + row * LSTR + 16 + hl * 8);  // K 16-23 / 24-31
            af[mi] = u.h;
        }
#pragma unroll
        for (int ni = 0; ni < 4; ++ni) {
            int nr = wn * 64 + ni * 16 + lid;
            FragU u;
            u.f4[0] = *(const float4*)(Bs + nr * LSTR + hl * 8);
            u.f4[1] = *(const float4*)(Bs + nr * LSTR + 16 + hl * 8);
            bf[ni] = u.h;
        }
#pragma unroll
        for (int mi = 0; mi < 4; ++mi)
#pragma unroll
            for (int ni = 0; ni < 4; ++ni)
                acc[mi][ni] = __builtin_amdgcn_wmma_f32_16x16x32_f16(
                    false, af[mi], false, bf[ni], (short)0, acc[mi][ni], false, false);
    };

    // K is always a multiple of 64: ping-pong with literal buffer ids
    issue(0, 0);
    asyncwait();
    __syncthreads();
    for (int k0 = 0; k0 < K; k0 += 2 * BK) {
        if (k0 + BK < K) issue(1, k0 + BK);
        compute(0);
        asyncwait();
        __syncthreads();
        if (k0 + 2 * BK < K) issue(0, k0 + 2 * BK);
        compute(1);
        asyncwait();
        __syncthreads();
    }

    // epilogue: C layout = VGPR j -> row base + hl*8 + j, col = base + lid
#pragma unroll
    for (int mi = 0; mi < 4; ++mi) {
#pragma unroll
        for (int ni = 0; ni < 4; ++ni) {
            int col = n0 + wn * 64 + ni * 16 + lid;
            if (col >= ldo) continue;
            float bc = 0.f;
            if (HASBIAS && col < N) bc = bias[col];
#pragma unroll
            for (int j = 0; j < 8; ++j) {
                int row = m0 + wm * 64 + mi * 16 + hl * 8 + j;
                if (row >= M) continue;
                float v = (col < N) ? (acc[mi][ni][j] + bc) : 0.f;
                if (OUT16) out16[(size_t)row * ldo + col] = (_Float16)v;
                else       out32[(size_t)row * ldo + col] = v;
            }
        }
    }
}

// ---------------------------------------------------------------------------
// Elementwise / reduction helpers
// ---------------------------------------------------------------------------
__global__ void k_convert_x(const float* __restrict__ x, _Float16* __restrict__ o,
                            int M, int K, int Kp) {
    int i = blockIdx.x * 256 + threadIdx.x;
    if (i >= M * Kp) return;
    int r = i / Kp, c = i % Kp;
    o[i] = (_Float16)((c < K) ? x[(size_t)r * K + c] : 0.f);
}

// W fp32 [K,N] -> f16 transposed+padded [N, Kp]
__global__ void k_convert_w(const float* __restrict__ W, _Float16* __restrict__ o,
                            int K, int N, int Kp) {
    int i = blockIdx.x * 256 + threadIdx.x;
    if (i >= N * Kp) return;
    int n = i / Kp, k = i % Kp;
    o[i] = (_Float16)((k < K) ? W[(size_t)k * N + n] : 0.f);
}

__global__ void k_fill(float* p, int n, float v) {
    int i = blockIdx.x * 256 + threadIdx.x;
    if (i < n) p[i] = v;
}

__global__ void k_deg(const int* __restrict__ dst, float* deg, int E) {
    int e = blockIdx.x * 256 + threadIdx.x;
    if (e < E) atomicAdd(&deg[dst[e]], 1.0f);
}

__global__ void k_rsqrt(float* p, int n) {
    int i = blockIdx.x * 256 + threadIdx.x;
    if (i < n) p[i] = rsqrtf(p[i]);
}

// column sum / sumsq of f16 buffer [M x ldo] (first N cols) -> st[0..] / st[1024..]
__global__ void k_colstats(const _Float16* __restrict__ b, int M, int N, int ldo,
                           int rows, float* st) {
    int c = blockIdx.x * 256 + threadIdx.x;
    if (c >= N) return;
    int r0 = blockIdx.y * rows;
    int r1 = min(M, r0 + rows);
    float s = 0.f, s2 = 0.f;
    for (int r = r0; r < r1; ++r) {
        float v = (float)b[(size_t)r * ldo + c];
        s += v; s2 += v * v;
    }
    atomicAdd(&st[c], s);
    atomicAdd(&st[1024 + c], s2);
}

__global__ void k_bn_fin(float* st, const float* __restrict__ g,
                         const float* __restrict__ beta, int N, float invM) {
    int c = blockIdx.x * 256 + threadIdx.x;
    if (c >= N) return;
    float mu  = st[c] * invM;
    float var = st[1024 + c] * invM - mu * mu;
    float inv = rsqrtf(var + 1e-5f);
    float sc  = g[c] * inv;
    st[2048 + c] = sc;
    st[3072 + c] = beta[c] - mu * sc;
}

// normalize + ReLU in place, 4 halves per thread (all N are multiples of 4)
__global__ void k_bn_relu(_Float16* b, int M, int N, int ldo, const float* __restrict__ st) {
    int nc = N >> 2;
    int i = blockIdx.x * 256 + threadIdx.x;
    if (i >= M * nc) return;
    int r = i / nc, c = (i % nc) << 2;
    size_t o = (size_t)r * ldo + c;
    H4 u; u.f = *(const float2*)(b + o);
#pragma unroll
    for (int j = 0; j < 4; ++j) {
        float v = (float)u.h[j];
        v = fmaxf(v * st[2048 + c + j] + st[3072 + c + j], 0.f);
        u.h[j] = (_Float16)v;
    }
    *(float2*)(b + o) = u.f;
}

// agg = bias + h * dinv^2 (self loop), float4 per thread (472 = 118*4)
__global__ void k_gcn_init(float* __restrict__ agg, const float* __restrict__ h,
                           const float* __restrict__ bias, const float* __restrict__ dinv,
                           int M) {
    int i = blockIdx.x * 256 + threadIdx.x;
    if (i >= M * 118) return;
    int r = i / 118, c = (i % 118) << 2;
    float d = dinv[r];
    float d2 = d * d;
    float4 hv = *(const float4*)(h + (size_t)r * 472 + c);
    float4 bv = *(const float4*)(bias + c);
    float4 o = make_float4(bv.x + hv.x * d2, bv.y + hv.y * d2,
                           bv.z + hv.z * d2, bv.w + hv.w * d2);
    *(float4*)(agg + (size_t)r * 472 + c) = o;
}

// agg[dst] += h[src] * dinv[src]*dinv[dst], float4 per thread
__global__ void k_gcn_scatter(float* __restrict__ agg, const float* __restrict__ h,
                              const int* __restrict__ src, const int* __restrict__ dst,
                              const float* __restrict__ dinv, int E) {
    int i = blockIdx.x * 256 + threadIdx.x;
    if (i >= E * 118) return;
    int e = i / 118, c = (i % 118) << 2;
    int s = src[e], d = dst[e];
    float w = dinv[s] * dinv[d];
    float4 hv = *(const float4*)(h + (size_t)s * 472 + c);
    float* o = agg + (size_t)d * 472 + c;
    atomicAdd(o + 0, hv.x * w);
    atomicAdd(o + 1, hv.y * w);
    atomicAdd(o + 2, hv.z * w);
    atomicAdd(o + 3, hv.w * w);
}

// f32 [M x 472] -> f16 [M x 512] with optional ReLU, 8 halves per thread
__global__ void k_relu_conv(const float* __restrict__ a, _Float16* __restrict__ o,
                            int M, int relu) {
    int i = blockIdx.x * 256 + threadIdx.x;
    if (i >= M * 64) return;
    int r = i / 64, c = (i % 64) << 3;
    H8 u;
    if (c < 472) {
        float4 lo = *(const float4*)(a + (size_t)r * 472 + c);
        float4 hi = *(const float4*)(a + (size_t)r * 472 + c + 4);
        float vv[8] = {lo.x, lo.y, lo.z, lo.w, hi.x, hi.y, hi.z, hi.w};
#pragma unroll
        for (int j = 0; j < 8; ++j) {
            float v = relu ? fmaxf(vv[j], 0.f) : vv[j];
            u.h[j] = (_Float16)v;
        }
    } else {
        u.f = make_float4(0.f, 0.f, 0.f, 0.f);
    }
    *(float4*)(o + (size_t)r * 512 + c) = u.f;
}

// edge gather: [E x 960] = concat(node[src][0:472], node[dst][0:472], pad); 8 halves/thread
__global__ void k_gather(const _Float16* __restrict__ nx, const int* __restrict__ src,
                         const int* __restrict__ dst, _Float16* __restrict__ o, int E) {
    int i = blockIdx.x * 256 + threadIdx.x;
    if (i >= E * 120) return;
    int e = i / 120, c = (i % 120) << 3;
    float4 v = make_float4(0.f, 0.f, 0.f, 0.f);
    if (c < 472)      v = *(const float4*)(nx + (size_t)src[e] * 512 + c);
    else if (c < 944) v = *(const float4*)(nx + (size_t)dst[e] * 512 + (c - 472));
    *(float4*)(o + (size_t)e * 960 + c) = v;
}

// final 944->1 dot + sigmoid, one wave per edge row
__global__ __launch_bounds__(256)
void k_dot_sig(const _Float16* __restrict__ b, const float* __restrict__ w,
               const float* __restrict__ bias, float* __restrict__ out, int E) {
    int wid  = (blockIdx.x * 256 + threadIdx.x) >> 5;
    int lane = threadIdx.x & 31;
    if (wid >= E) return;
    const _Float16* row = b + (size_t)wid * 960;
    float s = 0.f;
    for (int k = lane; k < 944; k += 32) s += (float)row[k] * w[k];
#pragma unroll
    for (int m = 16; m; m >>= 1) s += __shfl_xor(s, m, 32);
    if (lane == 0) out[wid] = 1.f / (1.f + expf(-(s + bias[0])));
}

// ---------------------------------------------------------------------------
extern "C" void kernel_launch(void* const* d_in, const int* in_sizes, int n_in,
                              void* d_out, int out_size, void* d_ws, size_t ws_size,
                              hipStream_t stream) {
    (void)in_sizes; (void)n_in; (void)out_size; (void)ws_size;
    const int NN = 20000, NE = 80000;
    const int F1 = 118, F2 = 236, F4 = 472, F8 = 944;
    const int P1 = 128, P2 = 256, P4 = 512, P8 = 960;

    const float* x   = (const float*)d_in[0];
    const int*   ei  = (const int*)d_in[1];
    const int*   src = ei;
    const int*   dst = ei + NE;
    const float* m1W[3] = {(const float*)d_in[2], (const float*)d_in[3], (const float*)d_in[4]};
    const float* m1b[3] = {(const float*)d_in[5], (const float*)d_in[6], (const float*)d_in[7]};
    const float* m1g[2] = {(const float*)d_in[8], (const float*)d_in[9]};
    const float* m1be[2]= {(const float*)d_in[10], (const float*)d_in[11]};
    const float* gW[4]  = {(const float*)d_in[12], (const float*)d_in[13], (const float*)d_in[14], (const float*)d_in[15]};
    const float* gb[4]  = {(const float*)d_in[16], (const float*)d_in[17], (const float*)d_in[18], (const float*)d_in[19]};
    const float* pW[4]  = {(const float*)d_in[20], (const float*)d_in[21], (const float*)d_in[22], (const float*)d_in[23]};
    const float* pb[4]  = {(const float*)d_in[24], (const float*)d_in[25], (const float*)d_in[26], (const float*)d_in[27]};
    const float* pg[3]  = {(const float*)d_in[28], (const float*)d_in[29], (const float*)d_in[30]};
    const float* pbe[3] = {(const float*)d_in[31], (const float*)d_in[32], (const float*)d_in[33]};

    // ---- workspace carving (256B aligned) ----
    size_t off = 0;
    char* base = (char*)d_ws;
    auto carve = [&](size_t bytes) -> char* {
        char* p = base + off;
        off += (bytes + 255) & ~(size_t)255;
        return p;
    };
    _Float16* wm1[3]; _Float16* wg[4]; _Float16* wp[3];
    wm1[0] = (_Float16*)carve((size_t)F2 * P1 * 2);
    wm1[1] = (_Float16*)carve((size_t)F2 * P2 * 2);
    wm1[2] = (_Float16*)carve((size_t)F2 * P2 * 2);
    wg[0]  = (_Float16*)carve((size_t)F4 * P2 * 2);
    for (int l = 1; l < 4; ++l) wg[l] = (_Float16*)carve((size_t)F4 * P4 * 2);
    for (int l = 0; l < 3; ++l) wp[l] = (_Float16*)carve((size_t)F8 * P8 * 2);
    _Float16* nA = (_Float16*)carve((size_t)NN * P4 * 2);
    _Float16* nB = (_Float16*)carve((size_t)NN * P4 * 2);
    float*    nF = (float*)   carve((size_t)NN * F4 * 4);
    float*    nG = (float*)   carve((size_t)NN * F4 * 4);
    _Float16* eA = (_Float16*)carve((size_t)NE * P8 * 2);
    _Float16* eB = (_Float16*)carve((size_t)NE * P8 * 2);
    float*    dinv  = (float*)carve((size_t)NN * 4);
    float*    stats = (float*)carve(4096 * 4);

    auto EL = [](int n) { return dim3((unsigned)((n + 255) / 256)); };

    auto gemm16 = [&](const _Float16* A, int lda, const _Float16* Wt, int ldw,
                      const float* bias, _Float16* O, int M, int N, int K, int ldo) {
        dim3 g((M + BM - 1) / BM, (ldo + BN - 1) / BN);
        k_gemm<true, true><<<g, 256, 0, stream>>>(A, lda, Wt, ldw, bias, O, nullptr, M, N, K, ldo);
    };
    auto gemm32 = [&](const _Float16* A, int lda, const _Float16* Wt, int ldw,
                      float* O, int M, int N, int K, int ldo) {
        dim3 g((M + BM - 1) / BM, (ldo + BN - 1) / BN);
        k_gemm<false, false><<<g, 256, 0, stream>>>(A, lda, Wt, ldw, nullptr, nullptr, O, M, N, K, ldo);
    };
    auto bnstage = [&](_Float16* buf, int M, int N, int ldo, const float* g, const float* be) {
        k_fill<<<EL(4096), 256, 0, stream>>>(stats, 4096, 0.f);
        const int rows = 256;
        dim3 gs((N + 255) / 256, (M + rows - 1) / rows);
        k_colstats<<<gs, 256, 0, stream>>>(buf, M, N, ldo, rows, stats);
        k_bn_fin<<<EL(1024), 256, 0, stream>>>(stats, g, be, N, 1.0f / (float)M);
        k_bn_relu<<<EL(M * (N >> 2)), 256, 0, stream>>>(buf, M, N, ldo, stats);
    };

    // ---- weight conversion (every call; deterministic) ----
    k_convert_w<<<EL(F2 * P1), 256, 0, stream>>>(m1W[0], wm1[0], F1, F2, P1);
    k_convert_w<<<EL(F2 * P2), 256, 0, stream>>>(m1W[1], wm1[1], F2, F2, P2);
    k_convert_w<<<EL(F2 * P2), 256, 0, stream>>>(m1W[2], wm1[2], F2, F2, P2);
    k_convert_w<<<EL(F4 * P2), 256, 0, stream>>>(gW[0], wg[0], F2, F4, P2);
    for (int l = 1; l < 4; ++l)
        k_convert_w<<<EL(F4 * P4), 256, 0, stream>>>(gW[l], wg[l], F4, F4, P4);
    for (int l = 0; l < 3; ++l)
        k_convert_w<<<EL(F8 * P8), 256, 0, stream>>>(pW[l], wp[l], F8, F8, P8);

    // ---- input + degree normalization ----
    k_convert_x<<<EL(NN * P1), 256, 0, stream>>>(x, nA, NN, F1, P1);
    k_fill<<<EL(NN), 256, 0, stream>>>(dinv, NN, 1.0f);           // self loop
    k_deg<<<EL(NE), 256, 0, stream>>>(dst, dinv, NE);
    k_rsqrt<<<EL(NN), 256, 0, stream>>>(dinv, NN);

    // ---- mlp1 ----
    gemm16(nA, P1, wm1[0], P1, m1b[0], nB, NN, F2, P1, P2);
    bnstage(nB, NN, F2, P2, m1g[0], m1be[0]);
    gemm16(nB, P2, wm1[1], P2, m1b[1], nA, NN, F2, P2, P2);
    bnstage(nA, NN, F2, P2, m1g[1], m1be[1]);
    gemm16(nA, P2, wm1[2], P2, m1b[2], nB, NN, F2, P2, P2);      // plain last

    // ---- gcn x4 ----
    _Float16* cin  = nB;  int cK = P2;
    _Float16* cout = nA;
    for (int l = 0; l < 4; ++l) {
        gemm32(cin, cK, wg[l], cK, nF, NN, F4, cK, F4);
        k_gcn_init<<<EL(NN * 118), 256, 0, stream>>>(nG, nF, gb[l], dinv, NN);
        k_gcn_scatter<<<EL(NE * 118), 256, 0, stream>>>(nG, nF, src, dst, dinv, NE);
        k_relu_conv<<<EL(NN * 64), 256, 0, stream>>>(nG, cout, NN, (l < 3) ? 1 : 0);
        _Float16* t = cin; cin = cout; cout = t;
        cK = P4;
    }
    _Float16* gout = cin;   // final gcn output f16 [NN x 512]

    // ---- edge gather + mlp2 ----
    k_gather<<<EL(NE * 120), 256, 0, stream>>>(gout, src, dst, eA, NE);
    gemm16(eA, P8, wp[0], P8, pb[0], eB, NE, F8, P8, P8);
    bnstage(eB, NE, F8, P8, pg[0], pbe[0]);
    gemm16(eB, P8, wp[1], P8, pb[1], eA, NE, F8, P8, P8);
    bnstage(eA, NE, F8, P8, pg[1], pbe[1]);
    gemm16(eA, P8, wp[2], P8, pb[2], eB, NE, F8, P8, P8);
    bnstage(eB, NE, F8, P8, pg[2], pbe[2]);

    // ---- final linear + sigmoid ----
    k_dot_sig<<<dim3((NE * 32 + 255) / 256), 256, 0, stream>>>(eB, pW[3], pb[3],
                                                               (float*)d_out, NE);
}
